// SelfLocalAttention_43602507989361
// MI455X (gfx1250) — matrix operand order repaired and checked by reading, not compiled, not run on hardware
//
#include <hip/hip_runtime.h>
#include <hip/hip_bf16.h>

#define BATCH 2
#define SEQ   2048
#define HIDN  512
#define NH    8
#define HD    64
#define WIN   128
#define MROWS (BATCH * SEQ)   // 4096

#define USE_ASYNC_LDS 1

typedef __bf16 bf16;
typedef __attribute__((ext_vector_type(16))) __bf16 v16bf;
typedef __attribute__((ext_vector_type(8)))  __bf16 v8bf;
typedef __attribute__((ext_vector_type(8)))  float  v8f;

static __device__ __forceinline__ v16bf cat8(v8bf lo, v8bf hi) {
  return __builtin_shufflevector(lo, hi, 0,1,2,3,4,5,6,7,8,9,10,11,12,13,14,15);
}

// Async DMA: copy 16B from global to LDS, tracked by ASYNCcnt (per-lane).
static __device__ __forceinline__ void async_copy_b128(const void* gptr, void* lptr) {
#if USE_ASYNC_LDS
  const unsigned int       lds = (unsigned int)(uintptr_t)lptr;   // generic->LDS offset
  const unsigned long long ga  = (unsigned long long)(uintptr_t)gptr;
  asm volatile("global_load_async_to_lds_b128 %0, %1, off" :: "v"(lds), "v"(ga) : "memory");
#else
  *(v8bf*)lptr = *(const v8bf*)gptr;
#endif
}
static __device__ __forceinline__ void async_wait() {
#if USE_ASYNC_LDS
  asm volatile("s_wait_asynccnt 0" ::: "memory");
#endif
}

// A-matrix fragment (16 rows x 32 K), row-major tile with `stride` elements/row.
// ISA layout: lanes 0-15 -> row = lane, K chunks {0..7,16..23};
//             lanes 16-31 -> row = lane-16, K chunks {8..15,24..31}.
static __device__ __forceinline__ v16bf frag_a(const bf16* base, int stride,
                                               int row_base, int col_base) {
  const int lane = threadIdx.x & 31;
  const bf16* p = base + (row_base + (lane & 15)) * stride + col_base + ((lane >> 4) << 3);
  return cat8(*(const v8bf*)p, *(const v8bf*)(p + 16));
}

// B-matrix fragment (32 K x 16 N) staged as [n][k] row-major tile.
// ISA layout: lane holds column N = lane%16, K chunks {0..7,16..23} (lanes 0-15) / {8..15,24..31} (lanes 16-31).
static __device__ __forceinline__ v16bf frag_b(const bf16* base, int stride,
                                               int row_base, int col_base) {
  const int lane = threadIdx.x & 31;
  const bf16* p = base + (row_base + (lane & 15)) * stride + col_base + ((lane >> 4) << 3);
  return cat8(*(const v8bf*)p, *(const v8bf*)(p + 16));
}

static __device__ __forceinline__ v8f wmma_bf16(v16bf a, v16bf b, v8f c) {
  return __builtin_amdgcn_wmma_f32_16x16x32_bf16(false, a, false, b, (short)0, c,
                                                 false, false);
}

// ---------------------------------------------------------------------------
// Tiled GEMM: Y[m][n] = sum_k A[m][k] * W[n][k] + bias[n]
// Tile 128(M) x 64(N), K-step 64; 8 waves, each wave: 16 rows x 64 cols (4 frags).
// OUT_MODE 0: fp32 row-major [M][HIDN]; OUT_MODE 1: bf16 head layout [B][NH][S][HD].
// ---------------------------------------------------------------------------
template <int OUT_MODE, bool A_BF16>
__global__ __launch_bounds__(256) void gemm_kernel(const void* __restrict__ Aglob,
                                                   const float* __restrict__ Wg,
                                                   const float* __restrict__ bias,
                                                   float* __restrict__ out_f32,
                                                   bf16* __restrict__ out_heads) {
  __shared__ bf16 As[128 * 72];   // 128 rows(m) x 64 k, pitch 72 (144B rows)
  __shared__ bf16 Bs[64 * 72];    // 64 rows(n)  x 64 k

  const int tid  = threadIdx.x;
  const int wave = tid >> 5;
  const int lane = tid & 31;
  const int m0 = blockIdx.x * 128;
  const int n0 = blockIdx.y * 64;
  const int wm = wave * 16;        // wave's 16-row strip

  v8f c0 = {}, c1 = {}, c2 = {}, c3 = {};

  // staging maps
  const int arow = tid >> 1;               // 0..127
  const int acol = (tid & 1) * 32;         // 0,32  (+ 4 chunks of 8)
  const int brow = tid >> 2;               // 0..63
  const int bcol = (tid & 3) * 16;         // 0,16,32,48 (+ 2 chunks of 8)

  for (int k0 = 0; k0 < HIDN; k0 += 64) {
    // ---- stage A tile (128 x 64) ----
    if (A_BF16) {
      const bf16* Ab = (const bf16*)Aglob + (size_t)(m0 + arow) * HIDN + k0 + acol;
      bf16* dst = &As[arow * 72 + acol];
#pragma unroll
      for (int c = 0; c < 4; ++c) async_copy_b128(Ab + c * 8, dst + c * 8);
    } else {
      const float* Af = (const float*)Aglob + (size_t)(m0 + arow) * HIDN + k0 + acol;
#pragma unroll
      for (int c = 0; c < 4; ++c) {
        v8bf av;
#pragma unroll
        for (int i = 0; i < 8; ++i) av[i] = (bf16)Af[c * 8 + i];
        *(v8bf*)&As[arow * 72 + acol + c * 8] = av;
      }
    }
    // ---- stage B tile: Bs[n][k] = W[n0+n][k0+k] (fp32 -> bf16) ----
    {
      const float* p = Wg + (size_t)(n0 + brow) * HIDN + k0 + bcol;
#pragma unroll
      for (int c = 0; c < 2; ++c) {
        v8bf bv;
#pragma unroll
        for (int i = 0; i < 8; ++i) bv[i] = (bf16)p[c * 8 + i];
        *(v8bf*)&Bs[brow * 72 + bcol + c * 8] = bv;
      }
    }
    async_wait();
    __syncthreads();

    const v16bf a0 = frag_a(As, 72, wm, 0);
    const v16bf a1 = frag_a(As, 72, wm, 32);
    c0 = wmma_bf16(a0, frag_b(Bs, 72, 0, 0), c0);
    c0 = wmma_bf16(a1, frag_b(Bs, 72, 0, 32), c0);
    c1 = wmma_bf16(a0, frag_b(Bs, 72, 16, 0), c1);
    c1 = wmma_bf16(a1, frag_b(Bs, 72, 16, 32), c1);
    c2 = wmma_bf16(a0, frag_b(Bs, 72, 32, 0), c2);
    c2 = wmma_bf16(a1, frag_b(Bs, 72, 32, 32), c2);
    c3 = wmma_bf16(a0, frag_b(Bs, 72, 48, 0), c3);
    c3 = wmma_bf16(a1, frag_b(Bs, 72, 48, 32), c3);
    __syncthreads();
  }

  // Epilogue: C layout -> lane holds rows r/(r+8) per lane half, col = lane%16
  const int hi  = lane >> 4;
  const int col = lane & 15;
  const float cc[4][8] = {{c0[0],c0[1],c0[2],c0[3],c0[4],c0[5],c0[6],c0[7]},
                          {c1[0],c1[1],c1[2],c1[3],c1[4],c1[5],c1[6],c1[7]},
                          {c2[0],c2[1],c2[2],c2[3],c2[4],c2[5],c2[6],c2[7]},
                          {c3[0],c3[1],c3[2],c3[3],c3[4],c3[5],c3[6],c3[7]}};
#pragma unroll
  for (int t = 0; t < 4; ++t) {
#pragma unroll
    for (int r = 0; r < 8; ++r) {
      const int row = m0 + wm + r + 8 * hi;
      const int cn  = n0 + t * 16 + col;
      const float y = cc[t][r] + bias[cn];
      if (OUT_MODE == 0) {
        out_f32[(size_t)row * HIDN + cn] = y;
      } else {
        const int b = row >> 11, s = row & (SEQ - 1);
        const int h = cn >> 6, d = cn & 63;
        out_heads[((((size_t)b * NH + h) * SEQ + s) << 6) + d] = (bf16)y;
      }
    }
  }
}

// ---------------------------------------------------------------------------
// Banded flash attention. Block = (b, h, 128-query tile), 8 waves x 16 queries.
// Loops over 12 key-chunks of 32 covering [qbase-128, qbase+255].
// ---------------------------------------------------------------------------
__global__ __launch_bounds__(256) void attn_kernel(const bf16* __restrict__ Qh,
                                                   const bf16* __restrict__ Kh,
                                                   const bf16* __restrict__ Vh,
                                                   bf16* __restrict__ Oout) {
  __shared__ bf16 Ks[32 * 72];     // [key][d]   pitch 72 elems (144B)
  __shared__ bf16 VsT[64 * 40];    // [d][key]   pitch 40 elems (80B)
  __shared__ bf16 Ps[8][16 * 40];  // per-wave P scratch, C-layout -> A-layout bridge

  const int tid  = threadIdx.x;
  const int wave = tid >> 5;
  const int lane = tid & 31;
  const int hi   = lane >> 4;
  const int col  = lane & 15;

  const int blk   = blockIdx.x;
  const int qblk  = blk & 15;
  const int h     = (blk >> 4) & 7;
  const int b     = blk >> 7;
  const int qbase = qblk * 128;
  const int q0    = qbase + wave * 16;

  const size_t headoff = (((size_t)b * NH + h) * SEQ) << 6;  // * HD
  const bf16* Qg = Qh + headoff;
  const bf16* Kg = Kh + headoff;
  const bf16* Vg = Vh + headoff;

  // Q fragments for both 32-wide K(d)-steps, loaded once (A-matrix pattern)
  v16bf qa0, qa1;
  {
    const bf16* p = Qg + ((size_t)(q0 + col) << 6) + (hi << 3);
    qa0 = cat8(*(const v8bf*)p, *(const v8bf*)(p + 16));
    qa1 = cat8(*(const v8bf*)(p + 32), *(const v8bf*)(p + 48));
  }

  float m[8], l[8];
  v8f o0 = {}, o1 = {}, o2 = {}, o3 = {};
#pragma unroll
  for (int r = 0; r < 8; ++r) { m[r] = -3.0e38f; l[r] = 0.0f; }

  const int kstart = qbase - WIN;
  const float scale = 0.125f;  // 1/sqrt(64)

  for (int jc = 0; jc < 12; ++jc) {
    const int kc = kstart + jc * 32;

    // Stage K (async DMA, row-major) and V (transposed, via VGPR); zero OOB keys
    {
      const int key = tid >> 3;          // 0..31
      const int d0  = (tid & 7) * 8;     // 0..56
      const int kg  = kc + key;
      if (kg >= 0 && kg < SEQ) {
        async_copy_b128(Kg + ((size_t)kg << 6) + d0, &Ks[key * 72 + d0]);
        const v8bf vv = *(const v8bf*)(Vg + ((size_t)kg << 6) + d0);
#pragma unroll
        for (int i = 0; i < 8; ++i) VsT[(d0 + i) * 40 + key] = vv[i];
      } else {
        const v8bf z = {};
        *(v8bf*)&Ks[key * 72 + d0] = z;
#pragma unroll
        for (int i = 0; i < 8; ++i) VsT[(d0 + i) * 40 + key] = (bf16)0.0f;
      }
    }
    async_wait();
    __syncthreads();

    // Scores: S = Q * K^T for 2 N-tiles of 16 keys, K(d) = 64 in two steps
    v8f s0 = {}, s1 = {};
    s0 = wmma_bf16(qa0, frag_b(Ks, 72, 0, 0), s0);
    s0 = wmma_bf16(qa1, frag_b(Ks, 72, 0, 32), s0);
    s1 = wmma_bf16(qa0, frag_b(Ks, 72, 16, 0), s1);
    s1 = wmma_bf16(qa1, frag_b(Ks, 72, 16, 32), s1);

    // Online softmax update (per lane: 8 row-partials; reduce across 16-lane halves)
    bf16* pp = &Ps[wave][0];
    const int key0 = kc + col;
    const int key1 = kc + 16 + col;
#pragma unroll
    for (int r = 0; r < 8; ++r) {
      const int q = q0 + r + 8 * hi;
      const float v0 = s0[r] * scale;
      const float v1 = s1[r] * scale;
      const int dq0 = q - key0, dq1 = q - key1;
      const bool ok0 = (key0 >= 0) && (key0 < SEQ) && (dq0 <= WIN) && (dq0 >= -WIN);
      const bool ok1 = (key1 >= 0) && (key1 < SEQ) && (dq1 <= WIN) && (dq1 >= -WIN);
      float mx = fmaxf(ok0 ? v0 : -3.0e38f, ok1 ? v1 : -3.0e38f);
#pragma unroll
      for (int off = 1; off < 16; off <<= 1) mx = fmaxf(mx, __shfl_xor(mx, off, 16));
      const float mnew = fmaxf(m[r], mx);
      const float f = __expf(m[r] - mnew);
      const float p0 = ok0 ? __expf(v0 - mnew) : 0.0f;
      const float p1 = ok1 ? __expf(v1 - mnew) : 0.0f;
      float rs = p0 + p1;
#pragma unroll
      for (int off = 1; off < 16; off <<= 1) rs += __shfl_xor(rs, off, 16);
      l[r] = l[r] * f + rs;
      m[r] = mnew;
      o0[r] *= f; o1[r] *= f; o2[r] *= f; o3[r] *= f;
      // C-layout -> LDS (becomes A-layout when re-read by frag_a)
      pp[(r + 8 * hi) * 40 + col]      = (bf16)p0;
      pp[(r + 8 * hi) * 40 + 16 + col] = (bf16)p1;
    }
    asm volatile("s_wait_dscnt 0" ::: "memory");  // same-wave LDS RAW turnaround

    // O += P * V  (4 d-tiles of 16)
    const v16bf pa = frag_a(pp, 40, 0, 0);
    o0 = wmma_bf16(pa, frag_b(VsT, 40, 0, 0), o0);
    o1 = wmma_bf16(pa, frag_b(VsT, 40, 16, 0), o1);
    o2 = wmma_bf16(pa, frag_b(VsT, 40, 32, 0), o2);
    o3 = wmma_bf16(pa, frag_b(VsT, 40, 48, 0), o3);
    __syncthreads();
  }

  // Normalize and write bf16 attention output in [B][S][HIDN] (GEMM-A layout)
#pragma unroll
  for (int r = 0; r < 8; ++r) {
    const float inv = 1.0f / l[r];
    const int row = q0 + r + 8 * hi;
    const size_t base = ((size_t)b * SEQ + row) * HIDN + h * HD + col;
    Oout[base]      = (bf16)(o0[r] * inv);
    Oout[base + 16] = (bf16)(o1[r] * inv);
    Oout[base + 32] = (bf16)(o2[r] * inv);
    Oout[base + 48] = (bf16)(o3[r] * inv);
  }
}

extern "C" void kernel_launch(void* const* d_in, const int* in_sizes, int n_in,
                              void* d_out, int out_size, void* d_ws, size_t ws_size,
                              hipStream_t stream) {
  (void)in_sizes; (void)n_in; (void)out_size; (void)ws_size;
  const float* value = (const float*)d_in[0];
  const float* key_  = (const float*)d_in[1];
  const float* query = (const float*)d_in[2];
  const float* Wq = (const float*)d_in[3];  const float* bq = (const float*)d_in[4];
  const float* Wk = (const float*)d_in[5];  const float* bk = (const float*)d_in[6];
  const float* Wv = (const float*)d_in[7];  const float* bv = (const float*)d_in[8];
  const float* Wo = (const float*)d_in[9];  const float* bo = (const float*)d_in[10];
  float* out = (float*)d_out;

  const size_t headElems = (size_t)BATCH * NH * SEQ * HD;  // 2,097,152
  bf16* Qh = (bf16*)d_ws;
  bf16* Kh = Qh + headElems;
  bf16* Vh = Kh + headElems;
  bf16* Ao = Vh + headElems;  // [MROWS][HIDN] bf16; total ws use = 16 MB

  dim3 gg(MROWS / 128, HIDN / 64);  // 32 x 8 blocks
  gemm_kernel<1, false><<<gg, 256, 0, stream>>>(query, Wq, bq, nullptr, Qh);
  gemm_kernel<1, false><<<gg, 256, 0, stream>>>(key_,  Wk, bk, nullptr, Kh);
  gemm_kernel<1, false><<<gg, 256, 0, stream>>>(value, Wv, bv, nullptr, Vh);

  attn_kernel<<<BATCH * NH * (SEQ / 128), 256, 0, stream>>>(Qh, Kh, Vh, Ao);

  gemm_kernel<0, true><<<gg, 256, 0, stream>>>(Ao, Wo, bo, out, nullptr);
}
